// Hankel_34411277976345
// MI455X (gfx1250) — matrix operand order repaired and checked by reading, not compiled, not run on hardware
//
#include <hip/hip_runtime.h>

typedef __attribute__((ext_vector_type(16))) _Float16 v16h;
typedef __attribute__((ext_vector_type(8)))  _Float16 v8h;
typedef __attribute__((ext_vector_type(4)))  _Float16 v4h;
typedef __attribute__((ext_vector_type(8)))  float    v8f;

// ---- problem constants (fixed by reference) ----
#define LSEQ   8
#define A_IN   16
#define O_IN   32
#define ENC    32
#define RANK   8

// Use CDNA5 async global->LDS copies for the block-shared B-fragment staging.
// Fallback (0): plain global_load_b128 + ds_store_b128.
#define ASYNC_LDS 1

// ---- prepped B-fragment layout in d_ws (units: halfs) ----
// Each fragment: [lane 0..31][16 halfs] = 512 halfs; lane loads 32 contiguous bytes.
#define OFF_WA     0                       // [2 nt][frag]           Wa^T (K=16 padded to 32)
#define OFF_WO     1024                    // [2 nt][frag]           Wo^T
#define OFF_FIRST  2048                    // [32 c][frag]           mps0    cols: l (8 valid)
#define OFF_MID    18432                   // [6][32 c][4 nt][frag]  mps_mid cols: n=i*8+l
#define OFF_LAST   411648                  // [32 c][frag]           mps_last cols: i (8 valid)
#define TOT_FRAG   428032                  // halfs (856,064 bytes of d_ws)

// ---- per-wave LDS layout (bytes); raw buffers overlay M/tmp (disjoint lifetime) ----
#define L_ENCA  0        // 8*16*32 halfs = 8192 B
#define L_ENCO  8192     // 8*16*32 halfs = 8192 B
#define L_RAWA  16384    // 8*16*16 halfs = 4096 B   (phases 1-2 only)
#define L_RAWO  20480    // 8*16*32 halfs = 8192 B   (phases 1-2 only)
#define L_M     16384    // 16*64 f32     = 4096 B   (phase 3, overlays RAWA)
#define L_TMP   20480    // 2*16*8 f32    = 1024 B   (phase 3, overlays RAWO)
#define WAVE_BYTES 28672

#define WAVES_PER_BLOCK 8
#define STAGE_OFF  (WAVES_PER_BLOCK * WAVE_BYTES)   // 229376
#define STAGE_BYTES 65536                            // half a mid core: [32 c][2 nt][frag]
#define LDS_TOTAL  (STAGE_OFF + STAGE_BYTES)         // 294912 B (< 320 KB/WGP)

__device__ __forceinline__ v16h splat16(_Float16 x) {
  v16h r;
#pragma unroll
  for (int i = 0; i < 16; ++i) r[i] = x;
  return r;
}

__device__ __forceinline__ v8f wmma16(v16h a, v16h b, v8f c) {
  return __builtin_amdgcn_wmma_f32_16x16x32_f16(false, a, false, b, (short)0, c,
                                                false, false);
}

// one 16-byte global->LDS staging copy
__device__ __forceinline__ void stage16(char* ldsPtr, const char* gbase, unsigned gofs) {
#if ASYNC_LDS
  asm volatile("global_load_async_to_lds_b128 %0, %1, %2 offset:0"
               :: "v"((unsigned)(unsigned long long)ldsPtr), "v"(gofs), "s"(gbase)
               : "memory");
#else
  *(uint4*)ldsPtr = *(const uint4*)(gbase + gofs);
#endif
}

__device__ __forceinline__ void wait_stage() {
#if ASYNC_LDS
  asm volatile("s_wait_asynccnt 0" ::: "memory");
#else
  asm volatile("s_wait_dscnt 0" ::: "memory");
#endif
}

// ---------------------------------------------------------------------------
// Prep kernel: transpose f32 weights into f16 WMMA B-fragments (swizzled).
// Fragment element (lane, h): column n = 16*nt + (lane&15), K = (lane&16)+h.
// ---------------------------------------------------------------------------
__global__ void prep_frags(const float* __restrict__ Wa, const float* __restrict__ Wo,
                           const float* __restrict__ mps0, const float* __restrict__ mps_mid,
                           const float* __restrict__ mps_last, _Float16* __restrict__ out) {
  for (int idx = blockIdx.x * blockDim.x + threadIdx.x; idx < TOT_FRAG;
       idx += gridDim.x * blockDim.x) {
    const int h    = idx & 15;
    const int lane = (idx >> 4) & 31;
    const int kk   = (lane & 16) + h;
    const int col  = lane & 15;
    float v = 0.0f;
    if (idx < OFF_WO) {
      const int nt = (idx >> 9) & 1;
      const int oc = nt * 16 + col;
      v = (kk < A_IN) ? Wa[oc * A_IN + kk] : 0.0f;
    } else if (idx < OFF_FIRST) {
      const int e  = idx - OFF_WO;
      const int nt = (e >> 9) & 1;
      const int oc = nt * 16 + col;
      v = Wo[oc * O_IN + kk];
    } else if (idx < OFF_MID) {
      const int e = idx - OFF_FIRST;
      const int c = (e >> 9) & 31;
      v = (col < RANK) ? mps0[(c * 32 + kk) * RANK + col] : 0.0f;
    } else if (idx < OFF_LAST) {
      const int e    = idx - OFF_MID;
      const int nt   = (e >> 9) & 3;
      const int c    = (e >> 11) & 31;
      const int core = e >> 16;
      const int n    = nt * 16 + col;
      const int i    = n >> 3, l = n & 7;
      v = mps_mid[(((core * RANK + i) * 32 + c) * 32 + kk) * RANK + l];
    } else {
      const int e = idx - OFF_LAST;
      const int c = (e >> 9) & 31;
      v = (col < RANK) ? mps_last[(col * 32 + c) * 32 + kk] : 0.0f;
    }
    out[idx] = (_Float16)v;
  }
}

// ---------------------------------------------------------------------------
// Main fused kernel: 8 waves/block, 1 wave = 16 batch rows. Middle-core
// B-fragments are staged block-wide into LDS via async-to-LDS copies.
// ---------------------------------------------------------------------------
__global__ void __launch_bounds__(256)
mps_wmma_kernel(const float* __restrict__ actions, const float* __restrict__ obss,
                const float* __restrict__ ba, const float* __restrict__ bo,
                const _Float16* __restrict__ frag, float* __restrict__ out, int nTiles) {
  const int tid  = threadIdx.x;
  const int lane = tid & 31;
  const int wave = tid >> 5;
  const int tileRaw = blockIdx.x * WAVES_PER_BLOCK + wave;
  const int tile = (tileRaw < nTiles) ? tileRaw : (nTiles - 1);  // keep barriers uniform
  const int n0 = tile * 16;

  extern __shared__ char smem[];
  char* wb = smem + wave * WAVE_BYTES;
  _Float16* encA = (_Float16*)(wb + L_ENCA);
  _Float16* encO = (_Float16*)(wb + L_ENCO);
  _Float16* rawA = (_Float16*)(wb + L_RAWA);
  _Float16* rawO = (_Float16*)(wb + L_RAWO);
  float*    Mld  = (float*)(wb + L_M);
  float*    tmpW = (float*)(wb + L_TMP);
  char*     stage = smem + STAGE_OFF;

  const int col  = lane & 15;
  const int rOff = (lane < 16) ? 0 : 8;
  const int b0   = (lane < 16) ? 0 : 16;

  // ---- Phase 1: raw f32 -> f16 LDS ----
  {
    const float4* pa = (const float4*)(actions + (size_t)n0 * (LSEQ * A_IN));
#pragma unroll 4
    for (int s = 0; s < 16; ++s) {
      const int v = s * 32 + lane;
      const float4 q = pa[v];
      const int f0 = v * 4, row = f0 >> 7, rem = f0 & 127, t = rem >> 4, i0 = rem & 15;
      v4h h; h[0] = (_Float16)q.x; h[1] = (_Float16)q.y; h[2] = (_Float16)q.z; h[3] = (_Float16)q.w;
      *(v4h*)(rawA + (t * 16 + row) * 16 + i0) = h;
    }
    const float4* po = (const float4*)(obss + (size_t)n0 * (LSEQ * O_IN));
#pragma unroll 4
    for (int s = 0; s < 32; ++s) {
      const int v = s * 32 + lane;
      const float4 q = po[v];
      const int f0 = v * 4, row = f0 >> 8, rem = f0 & 255, t = rem >> 5, k0 = rem & 31;
      v4h h; h[0] = (_Float16)q.x; h[1] = (_Float16)q.y; h[2] = (_Float16)q.z; h[3] = (_Float16)q.w;
      *(v4h*)(rawO + (t * 16 + row) * 32 + k0) = h;
    }
  }
  asm volatile("s_wait_dscnt 0" ::: "memory");

  const float baL = ba[col], baH = ba[16 + col];
  const float boL = bo[col], boH = bo[16 + col];
  const v8f z8 = {};

  // encoder B-fragments hoisted into registers (reused for all 8 timesteps)
  const v16h fWa0 = *(const v16h*)(frag + OFF_WA + (0 * 32 + lane) * 16);
  const v16h fWa1 = *(const v16h*)(frag + OFF_WA + (1 * 32 + lane) * 16);
  const v16h fWo0 = *(const v16h*)(frag + OFF_WO + (0 * 32 + lane) * 16);
  const v16h fWo1 = *(const v16h*)(frag + OFF_WO + (1 * 32 + lane) * 16);

  // ---- Phase 2: encoders via WMMA ----
  for (int t = 0; t < LSEQ; ++t) {
    v16h fa = {};
    {
      v8h alo = *(const v8h*)(rawA + (t * 16 + col) * 16);
      v8h ahi = *(const v8h*)(rawA + (t * 16 + col) * 16 + 8);
      if (lane < 16) {
#pragma unroll
        for (int i = 0; i < 8; ++i) { fa[i] = alo[i]; fa[8 + i] = ahi[i]; }  // K 16..31 stay zero
      }
    }
    v16h fo;
    {
      v8h o1 = *(const v8h*)(rawO + (t * 16 + col) * 32 + b0);
      v8h o2 = *(const v8h*)(rawO + (t * 16 + col) * 32 + b0 + 8);
#pragma unroll
      for (int i = 0; i < 8; ++i) { fo[i] = o1[i]; fo[8 + i] = o2[i]; }
    }
#pragma unroll
    for (int nt = 0; nt < 2; ++nt) {
      v8f da = wmma16(fa, nt ? fWa1 : fWa0, z8);
      const float biasA = nt ? baH : baL;
#pragma unroll
      for (int e = 0; e < 8; ++e)
        encA[(t * 16 + rOff + e) * 32 + nt * 16 + col] = (_Float16)(da[e] + biasA);

      v8f dob = wmma16(fo, nt ? fWo1 : fWo0, z8);
      const float biasO = nt ? boH : boL;
#pragma unroll
      for (int e = 0; e < 8; ++e)
        encO[(t * 16 + rOff + e) * 32 + nt * 16 + col] = (_Float16)(dob[e] + biasO);
    }
  }
  asm volatile("s_wait_dscnt 0" ::: "memory");

  auto loadOfrag = [&](int t) -> v16h {
    v8h o1 = *(const v8h*)(encO + (t * 16 + col) * 32 + b0);
    v8h o2 = *(const v8h*)(encO + (t * 16 + col) * 32 + b0 + 8);
    v16h f;
#pragma unroll
    for (int i = 0; i < 8; ++i) { f[i] = o1[i]; f[8 + i] = o2[i]; }
    return f;
  };

  int cur = 0;

  // ---- Phase 3a: first core (t=0) ----
  {
    const v16h fo = loadOfrag(0);
    v8f acc = {};
#pragma unroll 4
    for (int c = 0; c < 32; ++c) {
      const _Float16 aval = encA[(0 * 16 + col) * 32 + c];
      const v16h fA = fo * splat16(aval);
      const v16h bw = *(const v16h*)(frag + OFF_FIRST + (c * 32 + lane) * 16);
      acc = wmma16(fA, bw, acc);
    }
    if (col < RANK) {
#pragma unroll
      for (int e = 0; e < 8; ++e) tmpW[cur * 128 + (rOff + e) * 8 + col] = acc[e];
    }
    asm volatile("s_wait_dscnt 0" ::: "memory");
  }

  // ---- Phase 3b: middle cores, block-shared staged B-fragments ----
  for (int t = 1; t < LSEQ - 1; ++t) {
    const int core = t - 1;
    const v16h fo = loadOfrag(t);
    v8f acc[4] = {};
#pragma unroll
    for (int half = 0; half < 2; ++half) {
      __syncthreads();                     // everyone done consuming previous slice
      // cooperative stage of 64 KB: nt in {2*half, 2*half+1}, all 32 chunks
#pragma unroll
      for (int s = 0; s < 16; ++s) {
        const int m = tid + 256 * s;       // 16B chunk id, 0..4095
        const int c = m >> 7, rem = m & 127, ntL = rem >> 6, inner = rem & 63;
        const unsigned gofs =
            2u * (unsigned)(OFF_MID + core * 65536 + c * 2048 + (half * 2 + ntL) * 512 + inner * 8);
        stage16(stage + m * 16, (const char*)frag, gofs);
      }
      wait_stage();
      __syncthreads();                     // slice visible to all waves
#pragma unroll 2
      for (int c = 0; c < 32; ++c) {
        const _Float16 aval = encA[(t * 16 + col) * 32 + c];
        const v16h fA = fo * splat16(aval);
        const _Float16* sc = (const _Float16*)(stage + c * 2048) + lane * 16;
        acc[half * 2 + 0] = wmma16(fA, *(const v16h*)(sc),       acc[half * 2 + 0]);
        acc[half * 2 + 1] = wmma16(fA, *(const v16h*)(sc + 512), acc[half * 2 + 1]);
      }
    }
#pragma unroll
    for (int e = 0; e < 8; ++e) {
      Mld[(rOff + e) * 64 + col]      = acc[0][e];
      Mld[(rOff + e) * 64 + 16 + col] = acc[1][e];
      Mld[(rOff + e) * 64 + 32 + col] = acc[2][e];
      Mld[(rOff + e) * 64 + 48 + col] = acc[3][e];
    }
    asm volatile("s_wait_dscnt 0" ::: "memory");
    float res[4];
#pragma unroll
    for (int q = 0; q < 4; ++q) {
      const int idx = lane * 4 + q, row = idx >> 3, l = idx & 7;
      float s = 0.0f;
#pragma unroll
      for (int i = 0; i < 8; ++i)
        s += tmpW[cur * 128 + row * 8 + i] * Mld[row * 64 + i * 8 + l];
      res[q] = s;
    }
#pragma unroll
    for (int q = 0; q < 4; ++q) {
      const int idx = lane * 4 + q, row = idx >> 3, l = idx & 7;
      tmpW[(1 - cur) * 128 + row * 8 + l] = res[q];
    }
    asm volatile("s_wait_dscnt 0" ::: "memory");
    cur ^= 1;
  }

  // ---- Phase 3c: last core (t=7) ----
  {
    const v16h fo = loadOfrag(LSEQ - 1);
    v8f acc = {};
#pragma unroll 4
    for (int c = 0; c < 32; ++c) {
      const _Float16 aval = encA[((LSEQ - 1) * 16 + col) * 32 + c];
      const v16h fA = fo * splat16(aval);
      const v16h bw = *(const v16h*)(frag + OFF_LAST + (c * 32 + lane) * 16);
      acc = wmma16(fA, bw, acc);
    }
    if (col < RANK) {
#pragma unroll
      for (int e = 0; e < 8; ++e) Mld[(rOff + e) * 64 + col] = acc[e];
    }
    asm volatile("s_wait_dscnt 0" ::: "memory");
    if (lane < 16 && tileRaw < nTiles) {
      float s = 0.0f;
#pragma unroll
      for (int i = 0; i < 8; ++i)
        s += tmpW[cur * 128 + lane * 8 + i] * Mld[lane * 64 + i];
      out[n0 + lane] = s;
    }
  }
}

// ---------------------------------------------------------------------------
extern "C" void kernel_launch(void* const* d_in, const int* in_sizes, int n_in,
                              void* d_out, int out_size, void* d_ws, size_t ws_size,
                              hipStream_t stream) {
  const float* actions  = (const float*)d_in[0];
  const float* obss     = (const float*)d_in[1];
  const float* Wa       = (const float*)d_in[2];
  const float* ba       = (const float*)d_in[3];
  const float* Wo       = (const float*)d_in[4];
  const float* bo       = (const float*)d_in[5];
  const float* mps0     = (const float*)d_in[6];
  const float* mps_mid  = (const float*)d_in[7];
  const float* mps_last = (const float*)d_in[8];
  float* out = (float*)d_out;
  _Float16* frags = (_Float16*)d_ws;   // needs 856,064 bytes of scratch

  const int nBatch = in_sizes[0] / (LSEQ * A_IN);
  const int nTiles = nBatch / 16;
  const int nBlocks = (nTiles + WAVES_PER_BLOCK - 1) / WAVES_PER_BLOCK;

  prep_frags<<<512, 256, 0, stream>>>(Wa, Wo, mps0, mps_mid, mps_last, frags);
  mps_wmma_kernel<<<nBlocks, 256, LDS_TOTAL, stream>>>(actions, obss, ba, bo,
                                                       frags, out, nTiles);
}
